// DeepVIO_CDE_54571854463101
// MI455X (gfx1250) — matrix operand order repaired and checked by reading, not compiled, not run on hardware
//
#include <hip/hip_runtime.h>
#include <cstdint>
#include <cstddef>

// ---------------- problem constants ----------------
#define B_    256
#define T_    11
#define VF_   512
#define IF_   256
#define FLEN_ 768
#define HID_  256
#define FEAT_ 257
#define NEVAL_ 10
#define WFN_  (HID_ * FEAT_)   // 65792
#define DXP_  272              // FEAT padded to 17*16 for WMMA tiling

typedef unsigned short u16;
typedef __attribute__((ext_vector_type(16))) __bf16 bf16x16;
typedef __attribute__((ext_vector_type(8)))  float   f32x8;
typedef __attribute__((ext_vector_type(4)))  unsigned int u32x4;
typedef __attribute__((ext_vector_type(4)))  int     i32x4;
typedef __attribute__((ext_vector_type(8)))  int     i32x8;

union FragAB { int4 q[2]; bf16x16 v; };

// f32 -> bf16 (round-to-nearest-even), stored as u16
__device__ __forceinline__ u16 f2bf(float f) {
  unsigned u = __builtin_bit_cast(unsigned, f);
  u += 0x7FFFu + ((u >> 16) & 1u);
  return (u16)(u >> 16);
}

__device__ __forceinline__ float fast_tanh(float x) {
  float e = __expf(2.0f * x);                       // v_exp_f32
  return 1.0f - 2.0f * __builtin_amdgcn_rcpf(e + 1.0f);
}

// D = A(16x32 bf16) * B(32x16 bf16) + C(f32)  -> v_wmma_f32_16x16x32_bf16
__device__ __forceinline__ f32x8 wmma_bf16(FragAB a, FragAB b, f32x8 c) {
  return __builtin_amdgcn_wmma_f32_16x16x32_bf16(false, a.v, false, b.v,
                                                 (short)0, c, false, false);
}

// A fragment: row-major bf16 matrix (tile base = first row), per ISA layout:
// lane L: m=L&15, hi=L>>4; elems 0..7 -> K = k0+hi*8+e ; elems 8..15 -> K = k0+16+hi*8+(e-8)
__device__ __forceinline__ FragAB load_a(const u16* Abase, int lda, int k0) {
  int lane = threadIdx.x & 31;
  int m = lane & 15, hi = lane >> 4;
  const u16* p = Abase + m * lda + k0 + hi * 8;
  FragAB f;
  f.q[0] = *(const int4*)(p);
  f.q[1] = *(const int4*)(p + 16);
  return f;
}

// B fragment from a transposed weight (BT[n][k], row-major, ldb=K):
// lane L: n=L&15, hi=L>>4; elems 0..7 -> K = k0+hi*8+e ; elems 8..15 -> K = k0+16+hi*8+(e-8)
__device__ __forceinline__ FragAB load_bt(const u16* BT, int ldb, int k0) {
  int lane = threadIdx.x & 31;
  int n = lane & 15, hi = lane >> 4;
  const u16* p = BT + n * ldb + k0 + hi * 8;
  FragAB f;
  f.q[0] = *(const int4*)(p);
  f.q[1] = *(const int4*)(p + 16);
  return f;
}

// B fragment where every lane supplies its own row pointer (for clamped rows)
__device__ __forceinline__ FragAB load_bt_row(const u16* rowp, int k0, int hi) {
  const u16* p = rowp + k0 + hi * 8;
  FragAB f;
  f.q[0] = *(const int4*)(p);
  f.q[1] = *(const int4*)(p + 16);
  return f;
}

// ---------------- Tensor Data Mover: 2D tile -> LDS (CDNA5 ISA) ----------------
#if defined(__gfx1250__) && __has_builtin(__builtin_amdgcn_tensor_load_to_lds)
#define HAVE_TDM 1
// One TENSOR_LOAD_TO_LDS: tile_d0 x tile_d1 elements (data_size_code: 0=1B,1=2B,2=4B),
// row stride d0_stride elements; optional LDS row padding (interval/amount codes per ISA).
__device__ __forceinline__ void tdm_load_2d(unsigned lds_addr, unsigned long long gaddr,
                                            unsigned data_size_code,
                                            unsigned tile_d0, unsigned tile_d1,
                                            unsigned tensor_d0, unsigned tensor_d1,
                                            unsigned long long d0_stride,
                                            unsigned pad_en, unsigned pad_interval_code,
                                            unsigned pad_amount_code) {
  u32x4 g0;
  g0[0] = 1u;                                                     // count=1, user D#
  g0[1] = lds_addr;                                               // lds_addr [63:32]
  g0[2] = (unsigned)(gaddr & 0xFFFFFFFFull);                      // global_addr [95:64]
  g0[3] = (unsigned)((gaddr >> 32) & 0x01FFFFFFull) | (2u << 30); // addr[56:32] | type=2
  i32x8 g1;
  g1[0] = (int)((data_size_code << 16) | (pad_en << 20) |
                (pad_interval_code << 22) | (pad_amount_code << 25)); // wg_mask=0
  g1[1] = (int)((tensor_d0 & 0xFFFFu) << 16);                     // tensor_dim0[15:0]@48
  g1[2] = (int)(((tensor_d0 >> 16) & 0xFFFFu) | ((tensor_d1 & 0xFFFFu) << 16));
  g1[3] = (int)(((tensor_d1 >> 16) & 0xFFFFu) | ((tile_d0 & 0xFFFFu) << 16));
  g1[4] = (int)(tile_d1 & 0xFFFFu);                               // tile_dim1; tile_dim2=0
  g1[5] = (int)(unsigned)(d0_stride & 0xFFFFFFFFull);             // dim0_stride[31:0]@160
  g1[6] = (int)(unsigned)((d0_stride >> 32) & 0xFFFFull);         // dim0_stride[47:32]
  g1[7] = 0;
  i32x4 z4 = {0, 0, 0, 0};
#if __has_include(<hip/amd_detail/amd_gfx1250_TDM.h>)
  i32x8 z8 = {0, 0, 0, 0, 0, 0, 0, 0};
  __builtin_amdgcn_tensor_load_to_lds(g0, g1, z4, z4, z8, 0);     // 6-arg toolchain
#else
  __builtin_amdgcn_tensor_load_to_lds(g0, g1, z4, z4, 0);         // 5-arg ROCm 7.2
#endif
}
#else
#define HAVE_TDM 0
#endif

// ---------------- prep kernels ----------------

// LDS-tiled transpose + f32->bf16:  in (K x N, f32) -> out (N x K, bf16)
__global__ void transpose_bf16_kernel(const float* __restrict__ in,
                                      u16* __restrict__ out, int K, int N) {
  __shared__ float tile[16][17];
  int kb = blockIdx.x * 16, nb = blockIdx.y * 16;
  int tx = threadIdx.x & 15, ty = threadIdx.x >> 4;   // 256 threads
  tile[ty][tx] = in[(size_t)(kb + ty) * N + (nb + tx)];   // coalesced read
  __syncthreads();
  out[(size_t)(nb + ty) * K + (kb + tx)] = f2bf(tile[tx][ty]); // coalesced write
}

// XA = bf16(concat(fv, fi)) : (B*T) x 768
__global__ void xa_kernel(const float* __restrict__ fv, const float* __restrict__ fi,
                          u16* __restrict__ XA) {
  int idx = blockIdx.x * 256 + threadIdx.x;
  if (idx >= (B_ * T_) * FLEN_) return;
  int row = idx / FLEN_, c = idx % FLEN_;
  float v = (c < VF_) ? fv[row * VF_ + c] : fi[row * IF_ + (c - VF_)];
  XA[idx] = f2bf(v);
}

// Generic tiled WMMA GEMM: C(MxN,f32) = act(A(MxK,bf16) @ BT(NxK,bf16)^T + bias)
// 8 waves/block, one 16x16 tile per wave. act: 0=none, 1=leaky(0.1)
__global__ void gemm_bf16_kernel(const u16* __restrict__ A, const u16* __restrict__ BT,
                                 const float* __restrict__ bias, float* __restrict__ C,
                                 int K, int N, int act) {
  int wave = threadIdx.x >> 5;
  int tile = blockIdx.x * 8 + wave;
  int ntn = N >> 4;
  int rt = tile / ntn, ct = tile % ntn;
  const u16* Ab = A + (size_t)rt * 16 * K;
  const u16* Bb = BT + (size_t)ct * 16 * K;
  f32x8 acc = {0.f, 0.f, 0.f, 0.f, 0.f, 0.f, 0.f, 0.f};
  for (int k0 = 0; k0 < K; k0 += 32)
    acc = wmma_bf16(load_a(Ab, K, k0), load_bt(Bb, K, k0), acc);
  int lane = threadIdx.x & 31, n = lane & 15, hi = lane >> 4;
  float bv = bias[ct * 16 + n];
#pragma unroll
  for (int r = 0; r < 8; ++r) {
    float v = acc[r] + bv;
    if (act == 1) v = v > 0.f ? v : 0.1f * v;
    C[(size_t)(rt * 16 + r + 8 * hi) * N + ct * 16 + n] = v;
  }
}

// dX precompute: cubic-Hermite derivative samples for all 36 eval points.
// dXall[j][b][i] (i padded to 272, zeros beyond 256)
__global__ void dx_kernel(const float* __restrict__ fused, const float* __restrict__ ts,
                          float* __restrict__ dXall) {
  int j = blockIdx.x;                 // eval slot 0..35
  int b = blockIdx.y;
  int step = j >> 2, st = j & 3;
  float off = (st == 0) ? 0.f : (st == 3 ? 1.f : 0.5f);
  float tt = (float)step + off;
  int idx = (int)floorf(tt);
  if (idx > T_ - 2) idx = T_ - 2;
  if (idx < 0) idx = 0;
  float s = tt - (float)idx;
  float cx0 = 6.f * s * s - 6.f * s;
  float cd0 = 3.f * s * s - 4.f * s + 1.f;
  float cx1 = -6.f * s * s + 6.f * s;
  float cd1 = 3.f * s * s - 2.f * s;
  float* outp = dXall + ((size_t)j * B_ + b) * DXP_;
  for (int i = threadIdx.x; i < DXP_; i += blockDim.x) {
    float v = 0.f;
    if (i < FEAT_) {
      auto X = [&](int t) -> float {
        return (i == 0) ? ts[b * (T_ + 1) + t + 1]
                        : fused[(size_t)(b * T_ + t) * HID_ + (i - 1)];
      };
      float x0 = X(idx), x1 = X(idx + 1);
      float d0 = (idx == 0) ? (x1 - x0) : (x0 - X(idx - 1));
      float d1 = x1 - x0;
      v = cx0 * x0 + cd0 * d0 + cx1 * x1 + cd1 * d1;
    }
    outp[i] = v;
  }
}

__global__ void init_kernel(float* __restrict__ z, u16* __restrict__ h_all) {
  int e = blockIdx.x * 256 + threadIdx.x;          // 65536 elements
  z[e] = 0.f;
  int b = e >> 8, h = e & 255;
  h_all[(size_t)(b * NEVAL_) * HID_ + h] = 0;      // bf16 zero
}

// ---------------- per-stage MLP: z_eval -> relu -> relu -> h2 (bf16) ----------------
// One workgroup, 32 waves, 270KB LDS (CDNA5 allows 320KB/WG).
__global__ void mlp_kernel(const float* __restrict__ z, const float* __restrict__ kprev,
                           float alpha,
                           const u16* __restrict__ W1T, const float* __restrict__ b1,
                           const u16* __restrict__ W2T, const float* __restrict__ b2,
                           u16* __restrict__ h2g) {
  extern __shared__ u16 sm[];
  u16* zb  = sm;                 // 256 x 264 bf16 (padded rows, conflict-free A loads)
  u16* h1b = sm + 256 * 264;     // 256 x 264 bf16
  const int tid = threadIdx.x;

  for (int e = tid; e < HID_ * B_; e += 1024) {
    int row = e >> 8, col = e & 255;
    zb[row * 264 + col] = f2bf(z[e] + alpha * kprev[e]);
  }
  __syncthreads();

  const int wave = tid >> 5, lane = tid & 31;
  const int n = lane & 15, hi = lane >> 4;

  // phase 1: h1 = relu(z_eval @ W1 + b1)
  for (int t = wave * 8; t < wave * 8 + 8; ++t) {
    int rt = t >> 4, ct = t & 15;
    const u16* Ab = zb + rt * 16 * 264;
    const u16* Bb = W1T + ct * 16 * 256;
    f32x8 acc = {0.f, 0.f, 0.f, 0.f, 0.f, 0.f, 0.f, 0.f};
#pragma unroll
    for (int kk = 0; kk < 8; ++kk)
      acc = wmma_bf16(load_a(Ab, 264, kk * 32), load_bt(Bb, 256, kk * 32), acc);
    float bv = b1[ct * 16 + n];
#pragma unroll
    for (int r = 0; r < 8; ++r) {
      float v = acc[r] + bv;
      v = v > 0.f ? v : 0.f;
      h1b[(rt * 16 + r + 8 * hi) * 264 + ct * 16 + n] = f2bf(v);
    }
  }
  __syncthreads();

  // phase 2: h2 = relu(h1 @ W2 + b2) -> global bf16
  for (int t = wave * 8; t < wave * 8 + 8; ++t) {
    int rt = t >> 4, ct = t & 15;
    const u16* Ab = h1b + rt * 16 * 264;
    const u16* Bb = W2T + ct * 16 * 256;
    f32x8 acc = {0.f, 0.f, 0.f, 0.f, 0.f, 0.f, 0.f, 0.f};
#pragma unroll
    for (int kk = 0; kk < 8; ++kk)
      acc = wmma_bf16(load_a(Ab, 264, kk * 32), load_bt(Bb, 256, kk * 32), acc);
    float bv = b2[ct * 16 + n];
#pragma unroll
    for (int r = 0; r < 8; ++r) {
      float v = acc[r] + bv;
      v = v > 0.f ? v : 0.f;
      h2g[(size_t)(rt * 16 + r + 8 * hi) * 256 + ct * 16 + n] = f2bf(v);
    }
  }
}

// ---------------- the money kernel: fused GEMM + tanh + dX contraction ----------------
// vf[b,h'] = sum_i tanh( (h2 @ Wf)[b, h'*257+i] + bf[h'*257+i] ) * dX[b,i]
// grid = (16 b-tiles, 256 h'), 8 waves/WG. Each wave owns i-subtiles {w, w+8} (+{16} for w==0).
// LDS staging is done by the Tensor Data Mover (one wave issues two descriptor loads,
// hardware applies the +16B row padding), tracked with TENSORcnt.
__global__ void vf_kernel(const u16* __restrict__ h2g, const u16* __restrict__ WfT,
                          const float* __restrict__ bfb, const float* __restrict__ dX,
                          float* __restrict__ kout) {
  __shared__ __align__(16) u16 h2s[16 * 264];
  __shared__ __align__(16) float dxs[16 * DXP_];
  __shared__ float vfacc[16];
  const int tid = threadIdx.x;
  const int b0 = blockIdx.x * 16;
  const int hp = blockIdx.y;

  if (tid < 16) vfacc[tid] = 0.f;

#if HAVE_TDM
  if (tid < 32) {   // wave 0 only: TDM ignores EXEC, keep it to one wave-uniform branch
    // h2 tile: 16 rows x 256 bf16 (stride 256); LDS rows padded by 4 dwords (16B)
    // after every 128 dwords (512B = one row) -> matches the 264-u16 padded layout.
    tdm_load_2d((unsigned)(uintptr_t)h2s,
                (unsigned long long)(uintptr_t)(h2g + (size_t)b0 * 256),
                1u, /*tile*/ 256u, 16u, /*tensor*/ 256u, 256u, /*stride*/ 256ull,
                /*pad*/ 1u, 6u, 3u);
    // dX tile: 16 rows x 272 f32, contiguous -> one 4352-element linear row
    tdm_load_2d((unsigned)(uintptr_t)dxs,
                (unsigned long long)(uintptr_t)(dX + (size_t)b0 * DXP_),
                2u, /*tile*/ 4352u, 1u, /*tensor*/ 4352u, 1u, /*stride*/ 4352ull,
                /*pad*/ 0u, 0u, 0u);
    __builtin_amdgcn_s_wait_tensorcnt(0);
  }
#else
  { // manual staging fallback (host pass / toolchains without the TDM builtin)
    int row = tid >> 4, chunk = tid & 15;
    const int4* src = (const int4*)(h2g + (size_t)(b0 + row) * 256 + chunk * 16);
    int4* dst = (int4*)(h2s + row * 264 + chunk * 16);
    dst[0] = src[0];
    dst[1] = src[1];
    for (int e = tid; e < 16 * DXP_; e += 256)
      dxs[e] = dX[(size_t)(b0 + e / DXP_) * DXP_ + (e % DXP_)];
  }
#endif
  __syncthreads();

  const int wave = tid >> 5, lane = tid & 31;
  const int n = lane & 15, hi = lane >> 4;
  const bool has2 = (wave == 0);

  const int i0 = wave * 16 + n;           // < 128 + 16, always < 257
  const int i1 = (wave + 8) * 16 + n;     // < 256, always < 257
  const int i2 = 256 + n;                 // only n==0 is a real column
  const int row0 = hp * 257 + i0;
  const int row1 = hp * 257 + i1;
  const int row2 = hp * 257 + 256;        // clamped: dX pad zeroes kill garbage cols

  f32x8 acc0 = {0.f, 0.f, 0.f, 0.f, 0.f, 0.f, 0.f, 0.f};
  f32x8 acc1 = acc0, acc2 = acc0;
#pragma unroll
  for (int kk = 0; kk < 8; ++kk) {
    FragAB a = load_a(h2s, 264, kk * 32);
    acc0 = wmma_bf16(a, load_bt_row(WfT + (size_t)row0 * 256, kk * 32, hi), acc0);
    acc1 = wmma_bf16(a, load_bt_row(WfT + (size_t)row1 * 256, kk * 32, hi), acc1);
    if (has2)
      acc2 = wmma_bf16(a, load_bt_row(WfT + (size_t)row2 * 256, kk * 32, hi), acc2);
  }

  float bf0 = bfb[row0], bf1 = bfb[row1];
  float bf2 = has2 ? bfb[row2] : 0.f;
  float part[8];
#pragma unroll
  for (int r = 0; r < 8; ++r) {
    int m = r + 8 * hi;
    float p = fast_tanh(acc0[r] + bf0) * dxs[m * DXP_ + i0]
            + fast_tanh(acc1[r] + bf1) * dxs[m * DXP_ + i1];
    if (has2) p += fast_tanh(acc2[r] + bf2) * dxs[m * DXP_ + i2];
    part[r] = p;
  }
#pragma unroll
  for (int r = 0; r < 8; ++r) {
    float v = part[r];
    v += __shfl_xor(v, 1, 32);
    v += __shfl_xor(v, 2, 32);
    v += __shfl_xor(v, 4, 32);
    v += __shfl_xor(v, 8, 32);          // sum over the 16 N-lanes of each half
    if (n == 0) atomicAdd(&vfacc[r + 8 * hi], v);   // ds_add_f32
  }
  __syncthreads();
  if (tid < 16) kout[(size_t)(b0 + tid) * 256 + hp] = vfacc[tid];
}

// RK4 combine + append to h_all (bf16)
__global__ void zupdate_kernel(float* __restrict__ z, const float* __restrict__ k1,
                               const float* __restrict__ k2, const float* __restrict__ k3,
                               const float* __restrict__ k4, u16* __restrict__ h_all,
                               int tk) {
  int e = blockIdx.x * 256 + threadIdx.x;
  float zn = z[e] + (k1[e] + 2.f * k2[e] + 2.f * k3[e] + k4[e]) * (1.f / 6.f);
  z[e] = zn;
  int b = e >> 8, h = e & 255;
  h_all[(size_t)(b * NEVAL_ + tk + 1) * HID_ + h] = f2bf(zn);
}

// poses = r @ Wr2 + br2 (tiny: 2560x6, K=128)
__global__ void readout2_kernel(const float* __restrict__ r, const float* __restrict__ Wr2,
                                const float* __restrict__ br2, float* __restrict__ poses) {
  int idx = blockIdx.x * 256 + threadIdx.x;
  if (idx >= 2560 * 6) return;
  int row = idx / 6, c = idx % 6;
  float s = br2[c];
#pragma unroll 8
  for (int k = 0; k < 128; ++k) s += r[row * 128 + k] * Wr2[k * 6 + c];
  poses[idx] = s;
}

// ---------------- workspace layout (bytes) ----------------
static constexpr size_t OFF_WFT   = 0;                        // 65792*256 u16
static constexpr size_t OFF_W1T   = OFF_WFT   + 33685504;
static constexpr size_t OFF_W2T   = OFF_W1T   + 131072;
static constexpr size_t OFF_WREDT = OFF_W2T   + 131072;       // 256x768 u16
static constexpr size_t OFF_WR1T  = OFF_WREDT + 393216;       // 128x256 u16
static constexpr size_t OFF_XA    = OFF_WR1T  + 65536;        // 2816x768 u16
static constexpr size_t OFF_FUSED = OFF_XA    + 4325376;      // 2816x256 f32
static constexpr size_t OFF_DX    = OFF_FUSED + 2883584;      // 36x256x272 f32
static constexpr size_t OFF_H2    = OFF_DX    + 10027008;     // 256x256 u16
static constexpr size_t OFF_Z     = OFF_H2    + 131072;       // 256x256 f32
static constexpr size_t OFF_K0    = OFF_Z     + 262144;       // 4 k-buffers f32
static constexpr size_t OFF_HALL  = OFF_K0    + 1048576;      // 2560x256 u16
static constexpr size_t OFF_R     = OFF_HALL  + 1310720;      // 2560x128 f32

extern "C" void kernel_launch(void* const* d_in, const int* in_sizes, int n_in,
                              void* d_out, int out_size, void* d_ws, size_t ws_size,
                              hipStream_t stream) {
  (void)in_sizes; (void)n_in; (void)out_size; (void)ws_size;
  const float* fv    = (const float*)d_in[0];
  const float* fi    = (const float*)d_in[1];
  const float* ts    = (const float*)d_in[2];
  const float* W_red = (const float*)d_in[3];
  const float* b_red = (const float*)d_in[4];
  const float* W1    = (const float*)d_in[5];
  const float* b1    = (const float*)d_in[6];
  const float* W2    = (const float*)d_in[7];
  const float* b2    = (const float*)d_in[8];
  const float* Wf    = (const float*)d_in[9];
  const float* bfv   = (const float*)d_in[10];
  const float* Wr1   = (const float*)d_in[11];
  const float* br1   = (const float*)d_in[12];
  const float* Wr2   = (const float*)d_in[13];
  const float* br2   = (const float*)d_in[14];

  char* ws = (char*)d_ws;
  u16*   WfT   = (u16*)(ws + OFF_WFT);
  u16*   W1T   = (u16*)(ws + OFF_W1T);
  u16*   W2T   = (u16*)(ws + OFF_W2T);
  u16*   WredT = (u16*)(ws + OFF_WREDT);
  u16*   Wr1T  = (u16*)(ws + OFF_WR1T);
  u16*   XA    = (u16*)(ws + OFF_XA);
  float* fused = (float*)(ws + OFF_FUSED);
  float* dXall = (float*)(ws + OFF_DX);
  u16*   h2g   = (u16*)(ws + OFF_H2);
  float* z     = (float*)(ws + OFF_Z);
  float* kb[4] = {(float*)(ws + OFF_K0),            (float*)(ws + OFF_K0 + 262144),
                  (float*)(ws + OFF_K0 + 2*262144), (float*)(ws + OFF_K0 + 3*262144)};
  u16*   h_all = (u16*)(ws + OFF_HALL);
  float* rbuf  = (float*)(ws + OFF_R);

  // weight transposes + bf16 conversion (one pass; Wf stays L2-resident afterwards)
  transpose_bf16_kernel<<<dim3(256/16, WFN_/16), 256, 0, stream>>>(Wf, WfT, 256, WFN_);
  transpose_bf16_kernel<<<dim3(256/16, 256/16),  256, 0, stream>>>(W1, W1T, 256, 256);
  transpose_bf16_kernel<<<dim3(256/16, 256/16),  256, 0, stream>>>(W2, W2T, 256, 256);
  transpose_bf16_kernel<<<dim3(768/16, 256/16),  256, 0, stream>>>(W_red, WredT, 768, 256);
  transpose_bf16_kernel<<<dim3(256/16, 128/16),  256, 0, stream>>>(Wr1, Wr1T, 256, 128);

  // input concat -> bf16, reduce GEMM, dXdt tables, z init
  xa_kernel<<<(B_*T_*FLEN_ + 255)/256, 256, 0, stream>>>(fv, fi, XA);
  gemm_bf16_kernel<<<(176*16)/8, 256, 0, stream>>>(XA, WredT, b_red, fused, FLEN_, HID_, 0);
  dx_kernel<<<dim3(36, B_), 256, 0, stream>>>(fused, ts, dXall);
  init_kernel<<<(B_*HID_)/256, 256, 0, stream>>>(z, h_all);

  // opt in to >64KB dynamic LDS for the fused MLP (CDNA5: 320KB/WG)
  const size_t mlp_lds = (size_t)2 * 256 * 264 * sizeof(u16);   // 270336 B
  hipFuncSetAttribute(reinterpret_cast<const void*>(mlp_kernel),
                      hipFuncAttributeMaxDynamicSharedMemorySize, (int)mlp_lds);

  const float alphas[4] = {0.f, 0.5f, 0.5f, 1.f};
  for (int tk = 0; tk < 9; ++tk) {
    for (int st = 0; st < 4; ++st) {
      const float* kprev = (st == 0) ? z : kb[st - 1];
      mlp_kernel<<<1, 1024, mlp_lds, stream>>>(z, kprev, alphas[st], W1T, b1, W2T, b2, h2g);
      int j = tk * 4 + st;
      vf_kernel<<<dim3(B_/16, HID_), 256, 0, stream>>>(
          h2g, WfT, bfv, dXall + (size_t)j * B_ * DXP_, kb[st]);
    }
    zupdate_kernel<<<(B_*HID_)/256, 256, 0, stream>>>(z, kb[0], kb[1], kb[2], kb[3],
                                                      h_all, tk);
  }

  // readout: leaky(h_all @ Wr1 + br1) then @ Wr2 + br2
  gemm_bf16_kernel<<<(160*8)/8, 256, 0, stream>>>(h_all, Wr1T, br1, rbuf, HID_, 128, 1);
  readout2_kernel<<<(2560*6 + 255)/256, 256, 0, stream>>>(rbuf, Wr2, br2, (float*)d_out);

  // z_last -> second output slot
  hipMemcpyAsync((float*)d_out + B_ * NEVAL_ * 6, z, (size_t)B_ * HID_ * sizeof(float),
                 hipMemcpyDeviceToDevice, stream);
}
